// autoencoder_72756745994414
// MI455X (gfx1250) — matrix operand order repaired and checked by reading, not compiled, not run on hardware
//
#include <hip/hip_runtime.h>

// ---------------------------------------------------------------------------
// MI455X (gfx1250) autoencoder forward: 6x [GEMM(NT) + bias + LeakyReLU]
// bf16 WMMA (v_wmma_f32_16x16x32_bf16), fp32 accumulate.
// Block tile 128(M) x 256(N), 8 waves, 64x64 per wave (16 WMMA / K-step).
// K-tile staging via GLOBAL_LOAD_ASYNC_TO_LDS_B128 (ASYNCcnt) when the
// toolchain exposes the builtin; sync b128+ds_store fallback otherwise.
// ---------------------------------------------------------------------------

#define AS1 __attribute__((address_space(1)))
#define AS3 __attribute__((address_space(3)))

#if __has_builtin(__builtin_amdgcn_global_load_async_to_lds_b128) && \
    __has_builtin(__builtin_amdgcn_s_wait_asynccnt)
#define USE_ASYNC_LDS 1
#else
#define USE_ASYNC_LDS 0
#endif

typedef __attribute__((ext_vector_type(16))) __bf16 v16bf;
typedef __attribute__((ext_vector_type(8)))  __bf16 v8bf;
typedef __attribute__((ext_vector_type(8)))  float  v8f;
// Exact pointee type the async-LDS builtin expects (per hipcc diagnostic):
typedef int b128_t __attribute__((vector_size(4 * sizeof(int))));

#define TM   128   // workgroup M tile
#define TN   256   // workgroup N tile
#define TK   32    // K step (one WMMA K)
#define LDSS 40    // padded LDS row stride in bf16 elems (32 data + 8 pad)

// ---- fp32 -> bf16 conversion (8 elems / thread) ---------------------------
__global__ void __launch_bounds__(256)
cvt_f32_bf16(const float* __restrict__ in, __bf16* __restrict__ out, int n8) {
  int i = blockIdx.x * blockDim.x + threadIdx.x;
  if (i >= n8) return;
  const float4* p = (const float4*)in + (size_t)i * 2;
  float4 a = p[0];
  float4 b = p[1];
  v8bf o;
  o[0] = (__bf16)a.x; o[1] = (__bf16)a.y; o[2] = (__bf16)a.z; o[3] = (__bf16)a.w;
  o[4] = (__bf16)b.x; o[5] = (__bf16)b.y; o[6] = (__bf16)b.z; o[7] = (__bf16)b.w;
  *(v8bf*)(out + (size_t)i * 8) = o;
}

// ---- 16-byte tile-chunk staging: global -> LDS ----------------------------
__device__ __forceinline__ void stage16(const __bf16* g, __bf16* l) {
#if USE_ASYNC_LDS
  __builtin_amdgcn_global_load_async_to_lds_b128((AS1 b128_t*)g,
                                                 (AS3 b128_t*)l, 0, 0);
#else
  *(v8bf*)l = *(const v8bf*)g;
#endif
}

__device__ __forceinline__ void stage_wait() {
#if USE_ASYNC_LDS
  __builtin_amdgcn_s_wait_asynccnt(0);
#endif
}

// ---- fused GEMM(NT) + bias + LeakyReLU ------------------------------------
// A: MxK bf16 row-major (activations), Wt: NxK bf16 row-major (torch Linear
// weight), Out: MxN (bf16 or f32). C[m,n] = sum_k A[m,k]*Wt[n,k] + bias[n].
// M is fixed at 4096 (batch); N,K are compile-time per layer.
template <int N, int K, bool OUTF32>
__global__ void __launch_bounds__(256)
gemm_bias_lrelu(const __bf16* __restrict__ A, const __bf16* __restrict__ Wt,
                const float* __restrict__ bias, void* __restrict__ Out)
{
  __shared__ __align__(16) __bf16 sA[2][TM * LDSS];
  __shared__ __align__(16) __bf16 sW[2][TN * LDSS];

  const int tid  = threadIdx.x;
  const int lane = tid & 31;
  const int wave = tid >> 5;   // 8 waves
  const int wm   = wave & 1;   // 2 waves along M (64 rows each)
  const int wn   = wave >> 1;  // 4 waves along N (64 cols each)
  const int m0   = blockIdx.y * TM;
  const int n0   = blockIdx.x * TN;

  const __bf16* Ab = A  + (size_t)m0 * K;
  const __bf16* Wb = Wt + (size_t)n0 * K;

  // Tile staging: chunks of 8 bf16 (16B). A: 128x32 = 512 chunks (2/thread),
  // W: 256x32 = 1024 chunks (4/thread). 256 % 4 == 0 -> same K-quarter q.
  const int r = tid >> 2;         // base row 0..63
  const int q = (tid & 3) * 8;    // K-quarter offset in elems

  auto stage = [&](int buf, int kk) {
#pragma unroll
    for (int j = 0; j < 2; ++j)
      stage16(Ab + (size_t)(r + j * 64) * K + kk + q,
              &sA[buf][(r + j * 64) * LDSS + q]);
#pragma unroll
    for (int j = 0; j < 4; ++j)
      stage16(Wb + (size_t)(r + j * 64) * K + kk + q,
              &sW[buf][(r + j * 64) * LDSS + q]);
  };

  v8f acc[4][4];
  const v8f vzero = {0.f, 0.f, 0.f, 0.f, 0.f, 0.f, 0.f, 0.f};
#pragma unroll
  for (int i = 0; i < 4; ++i)
#pragma unroll
    for (int j = 0; j < 4; ++j)
      acc[i][j] = vzero;

  // Per-ISA 16-bit fragment striping: lanes 0-15 take K[0:8)+K[16:24) for A
  // and B, lanes 16-31 take K[8:16)+K[24:32) for A and B.
  const int koA = (lane & 16) ? 8 : 0;
  const int koB = (lane & 16) ? 8 : 0;
  const int l15 = lane & 15;

  stage(0, 0);
  stage_wait();
  __syncthreads();

  const int nK = K / TK;
  for (int ks = 0; ks < nK; ++ks) {
    const int  buf  = ks & 1;
    const bool more = (ks + 1) < nK;

    if (more) stage(buf ^ 1, (ks + 1) * TK);  // overlaps with WMMA below

    // A fragments: 4 M-subtiles of 16 rows for this wave.
    v16bf af[4];
#pragma unroll
    for (int mf = 0; mf < 4; ++mf) {
      const __bf16* p = &sA[buf][(wm * 64 + mf * 16 + l15) * LDSS + koA];
      v8bf lo = *(const v8bf*)p;
      v8bf hi = *(const v8bf*)(p + 16);
      af[mf] = __builtin_shufflevector(lo, hi, 0, 1, 2, 3, 4, 5, 6, 7,
                                       8, 9, 10, 11, 12, 13, 14, 15);
    }

#pragma unroll
    for (int nf = 0; nf < 4; ++nf) {
      const __bf16* p = &sW[buf][(wn * 64 + nf * 16 + l15) * LDSS + koB];
      v8bf lo = *(const v8bf*)p;
      v8bf hi = *(const v8bf*)(p + 16);
      v16bf bm = __builtin_shufflevector(lo, hi, 0, 1, 2, 3, 4, 5, 6, 7,
                                         8, 9, 10, 11, 12, 13, 14, 15);
#pragma unroll
      for (int mf = 0; mf < 4; ++mf)
        acc[mf][nf] = __builtin_amdgcn_wmma_f32_16x16x32_bf16(
            false, af[mf], false, bm, (short)0, acc[mf][nf], false, false);
    }

    if (more) {
      stage_wait();     // our async chunks for tile ks+1 are in LDS
      __syncthreads();  // everyone's chunks are in LDS; reads of buf done
    }
  }

  // Epilogue: bias + LeakyReLU, per C layout (lane -> N, VGPR idx -> M row).
#pragma unroll
  for (int nf = 0; nf < 4; ++nf) {
    const int   col = n0 + wn * 64 + nf * 16 + l15;
    const float bv  = bias[col];
#pragma unroll
    for (int mf = 0; mf < 4; ++mf) {
      const int rbase = m0 + wm * 64 + mf * 16 + ((lane & 16) ? 8 : 0);
      size_t    idx   = (size_t)rbase * N + col;
#pragma unroll
      for (int rr = 0; rr < 8; ++rr, idx += N) {
        float v = acc[mf][nf][rr] + bv;
        v = (v >= 0.0f) ? v : 0.01f * v;
        if (OUTF32) ((float*)Out)[idx]  = v;
        else        ((__bf16*)Out)[idx] = (__bf16)v;
      }
    }
  }
}

// ---------------------------------------------------------------------------
extern "C" void kernel_launch(void* const* d_in, const int* in_sizes, int n_in,
                              void* d_out, int out_size, void* d_ws, size_t ws_size,
                              hipStream_t stream) {
  (void)in_sizes; (void)n_in; (void)out_size; (void)ws_size;

  const float* x  = (const float*)d_in[0];
  const float* ew[3] = {(const float*)d_in[1], (const float*)d_in[2], (const float*)d_in[3]};
  const float* eb[3] = {(const float*)d_in[4], (const float*)d_in[5], (const float*)d_in[6]};
  const float* dw[3] = {(const float*)d_in[7], (const float*)d_in[8], (const float*)d_in[9]};
  const float* db[3] = {(const float*)d_in[10], (const float*)d_in[11], (const float*)d_in[12]};

  char* ws = (char*)d_ws;
  // Activation ping-pong (max 4096x4096 bf16 = 32 MB each).
  __bf16* buf0 = (__bf16*)(ws);
  __bf16* buf1 = (__bf16*)(ws + ((size_t)32 << 20));
  // bf16 weights.
  __bf16* wb1 = (__bf16*)(ws + ((size_t)64 << 20));       // 4096x1024
  __bf16* wb2 = wb1 + (size_t)4096 * 1024;                // 1024x4096
  __bf16* wb3 = wb2 + (size_t)1024 * 4096;                // 256x1024
  __bf16* wb4 = wb3 + (size_t)256 * 1024;                 // 1024x256
  __bf16* wb5 = wb4 + (size_t)1024 * 256;                 // 4096x1024
  __bf16* wb6 = wb5 + (size_t)4096 * 1024;                // 1024x4096

  auto cvt = [&](const float* src, __bf16* dst, size_t n) {
    int n8 = (int)(n / 8);
    cvt_f32_bf16<<<dim3((n8 + 255) / 256), dim3(256), 0, stream>>>(src, dst, n8);
  };

  // fp32 -> bf16 staging.
  cvt(x,     buf0, (size_t)4096 * 1024);
  cvt(ew[0], wb1,  (size_t)4096 * 1024);
  cvt(ew[1], wb2,  (size_t)1024 * 4096);
  cvt(ew[2], wb3,  (size_t)256  * 1024);
  cvt(dw[0], wb4,  (size_t)1024 * 256);
  cvt(dw[1], wb5,  (size_t)4096 * 1024);
  cvt(dw[2], wb6,  (size_t)1024 * 4096);

  const dim3 blk(256);
  // Encoder.
  gemm_bias_lrelu<4096, 1024, false><<<dim3(4096 / TN, 32), blk, 0, stream>>>(buf0, wb1, eb[0], buf1);
  gemm_bias_lrelu<1024, 4096, false><<<dim3(1024 / TN, 32), blk, 0, stream>>>(buf1, wb2, eb[1], buf0);
  gemm_bias_lrelu<256,  1024, false><<<dim3(256  / TN, 32), blk, 0, stream>>>(buf0, wb3, eb[2], buf1);
  // Decoder.
  gemm_bias_lrelu<1024, 256,  false><<<dim3(1024 / TN, 32), blk, 0, stream>>>(buf1, wb4, db[0], buf0);
  gemm_bias_lrelu<4096, 1024, false><<<dim3(4096 / TN, 32), blk, 0, stream>>>(buf0, wb5, db[1], buf1);
  gemm_bias_lrelu<1024, 4096, true ><<<dim3(1024 / TN, 32), blk, 0, stream>>>(buf1, wb6, db[2], d_out);
}
